// GraphNN_35562329210931
// MI455X (gfx1250) — compile-verified, matrix operand order checked by reading, not run on hardware
//
#include <hip/hip_runtime.h>

typedef __attribute__((ext_vector_type(16))) _Float16 v16h;
typedef __attribute__((ext_vector_type(8)))  _Float16 v8h;
typedef __attribute__((ext_vector_type(8)))  float    v8f;

#define TPB 256

__global__ void k_fill(float* __restrict__ p, float v, size_t n) {
  size_t i = (size_t)blockIdx.x * blockDim.x + threadIdx.x;
  if (i < n) p[i] = v;
}

__global__ void k_cvt_f16(const float* __restrict__ src, _Float16* __restrict__ dst, size_t n) {
  size_t i = (size_t)blockIdx.x * blockDim.x + threadIdx.x;
  if (i < n) dst[i] = (_Float16)src[i];
}

// W[k][c] (row-major KxCout) -> Wt[c][k] (row-major CoutxK), f32 -> f16
__global__ void k_cvt_wT(const float* __restrict__ W, _Float16* __restrict__ Wt,
                         int K, int Cout) {
  long gid = (long)blockIdx.x * blockDim.x + threadIdx.x;
  if (gid >= (long)K * Cout) return;
  int k = (int)(gid / Cout), c = (int)(gid % Cout);
  Wt[(size_t)c * K + k] = (_Float16)W[gid];
}

__global__ void k_deg_accum(const int* __restrict__ dst, float* __restrict__ deg, int nE) {
  int i = blockIdx.x * blockDim.x + threadIdx.x;
  if (i < nE) atomicAdd(&deg[dst[i]], 1.0f);
}

__global__ void k_rsqrt_inplace(float* __restrict__ p, int n) {
  int i = blockIdx.x * blockDim.x + threadIdx.x;
  if (i < n) p[i] = rsqrtf(p[i]);
}

// Hout[rows,Cout] = Ah[rows,K] @ Bt[Cout,K]^T (+bias if non-null).
// Ah: f16 row-major [rows][K]; Bt: f16 transposed weights [Cout][K].
// f16 WMMA, fp32 accumulate. rows%16==0, K%32==0, Cout%16==0.
// block=(32,8): 8 wave32 waves; wave ty owns tile (blockIdx.x, blockIdx.y*8+ty).
// Per K-step per lane: 2 b128 A-loads + 2 b128 B-loads + 1 v_wmma.
__global__ void k_gemm_wmma(const _Float16* __restrict__ Ah, const _Float16* __restrict__ Bt,
                            const float* __restrict__ bias, float* __restrict__ Hout,
                            int K, int Cout) {
  const int lane = threadIdx.x;           // 0..31 (wave32)
  const int m    = lane & 15;             // A row / B,D column within tile
  const int koff = (lane >> 4) * 8;       // K sub-offset for upper half-wave
  const int rowBase = blockIdx.x * 16;
  const int colBase = (blockIdx.y * 8 + threadIdx.y) * 16;

  // ISA 7.12.2 16-bit fragment layout: lanes 0-15 hold K 0..7 & 16..23,
  // lanes 16-31 hold K 8..15 & 24..31 (of each 32-wide K step).
  const _Float16* arow = Ah + (size_t)(rowBase + m) * K + koff;
  const _Float16* brow = Bt + (size_t)(colBase + m) * K + koff;

  v8f acc = {};
  for (int k0 = 0; k0 < K; k0 += 32) {
    v8h a0 = *(const v8h*)(arow + k0);
    v8h a1 = *(const v8h*)(arow + k0 + 16);
    v8h b0 = *(const v8h*)(brow + k0);
    v8h b1 = *(const v8h*)(brow + k0 + 16);
    v16h a = __builtin_shufflevector(a0, a1, 0, 1, 2, 3, 4, 5, 6, 7,
                                     8, 9, 10, 11, 12, 13, 14, 15);
    v16h b = __builtin_shufflevector(b0, b1, 0, 1, 2, 3, 4, 5, 6, 7,
                                     8, 9, 10, 11, 12, 13, 14, 15);
    acc = __builtin_amdgcn_wmma_f32_16x16x32_f16(false, a, false, b, (short)0, acc,
                                                 false, false);
  }

  // D layout: VGPR r -> row r (lanes 0-15) / row r+8 (lanes 16-31), col = lane&15
  const int col = colBase + m;
  const float bv = bias ? bias[col] : 0.0f;
#pragma unroll
  for (int r = 0; r < 8; ++r)
    Hout[(size_t)(rowBase + koff + r) * Cout + col] = acc[r] + bv;
}

// out[dst] += hw[src] * dinv[src]*dinv[dst], 4 channels per thread (float4 load).
__global__ void k_edge_scatter(const float* __restrict__ hw, const int* __restrict__ src,
                               const int* __restrict__ dst, const float* __restrict__ dinv,
                               float* __restrict__ out, int nE, int C) {
  long gid = (long)blockIdx.x * blockDim.x + threadIdx.x;
  int lpe = C >> 2;                      // threads per edge
  long e = gid / lpe;
  if (e >= nE) return;
  int c = (int)(gid % lpe) << 2;
  int s = src[e], d = dst[e];
  float en = dinv[s] * dinv[d];
  float4 v = *(const float4*)(hw + (size_t)s * C + c);
  float* o = out + (size_t)d * C + c;
  atomicAdd(o + 0, v.x * en);
  atomicAdd(o + 1, v.y * en);
  atomicAdd(o + 2, v.z * en);
  atomicAdd(o + 3, v.w * en);
}

// A[i,c] += B[i,c]*dinv[i]^2 + bias[c]   (self-loop term + bias)
__global__ void k_node_update(float* __restrict__ A, const float* __restrict__ B,
                              const float* __restrict__ dinv, const float* __restrict__ bias,
                              int n, int C) {
  long gid = (long)blockIdx.x * blockDim.x + threadIdx.x;
  if (gid >= (long)n * C) return;
  int i = (int)(gid / C), c = (int)(gid % C);
  float sn = dinv[i];
  A[gid] += B[gid] * sn * sn + bias[c];
}

__global__ void k_relu(float* __restrict__ A, size_t n) {
  size_t i = (size_t)blockIdx.x * blockDim.x + threadIdx.x;
  if (i < n) A[i] = fmaxf(A[i], 0.0f);
}

// per-column sum & sumsq; blockDim.x == C, each block covers 64 rows.
__global__ void k_colstats(const float* __restrict__ A, float* __restrict__ stats,
                           int n, int C) {
  int c = threadIdx.x;
  int r0 = blockIdx.x * 64;
  int r1 = r0 + 64; if (r1 > n) r1 = n;
  float s = 0.0f, s2 = 0.0f;
  for (int r = r0; r < r1; ++r) {
    float v = A[(size_t)r * C + c];
    s += v; s2 += v * v;
  }
  atomicAdd(&stats[c], s);
  atomicAdd(&stats[C + c], s2);
}

// A = relu((A - mean) * rsqrt(var+eps) * gamma + beta), biased var
__global__ void k_bn_relu(float* __restrict__ A, const float* __restrict__ stats,
                          const float* __restrict__ gamma, const float* __restrict__ beta,
                          int n, int C) {
  long gid = (long)blockIdx.x * blockDim.x + threadIdx.x;
  if (gid >= (long)n * C) return;
  int c = (int)(gid % C);
  float invN = 1.0f / (float)n;
  float m = stats[c] * invN;
  float var = stats[C + c] * invN - m * m;
  float is = rsqrtf(var + 1e-5f);
  float v = (A[gid] - m) * is * gamma[c] + beta[c];
  A[gid] = fmaxf(v, 0.0f);
}

__global__ void k_pool(const float* __restrict__ h, const int* __restrict__ batch,
                       float* __restrict__ hp, int n, int C) {
  long gid = (long)blockIdx.x * blockDim.x + threadIdx.x;
  if (gid >= (long)n * C) return;
  int i = (int)(gid / C), c = (int)(gid % C);
  atomicAdd(&hp[(size_t)batch[i] * C + c], h[gid]);
}

// out[g,o] = h[g,:] @ Wout[:,o] + bout[o]   (Cout = 2)
__global__ void k_out(const float* __restrict__ h, const float* __restrict__ Wout,
                      const float* __restrict__ bout, float* __restrict__ out,
                      int G, int K) {
  int gid = blockIdx.x * blockDim.x + threadIdx.x;
  if (gid >= G * 2) return;
  int g = gid >> 1, o = gid & 1;
  float s = bout[o];
  const float* hr = h + (size_t)g * K;
  for (int k = 0; k < K; ++k) s += hr[k] * Wout[k * 2 + o];
  out[gid] = s;
}

extern "C" void kernel_launch(void* const* d_in, const int* in_sizes, int n_in,
                              void* d_out, int out_size, void* d_ws, size_t ws_size,
                              hipStream_t stream) {
  (void)n_in; (void)ws_size;
  const float* x     = (const float*)d_in[0];
  const int*   eidx  = (const int*)d_in[1];
  const int*   batch = (const int*)d_in[2];
  // d_in[3] = num_graphs (device scalar; G derived from out_size instead)
  const float* W0    = (const float*)d_in[4];
  const float* b0    = (const float*)d_in[5];
  const float* Wg    = (const float*)d_in[6];
  const float* bg    = (const float*)d_in[7];
  const float* gam_g = (const float*)d_in[8];
  const float* bet_g = (const float*)d_in[9];
  const float* Wl0   = (const float*)d_in[10];
  const float* bl0   = (const float*)d_in[11];
  const float* Wl    = (const float*)d_in[12];
  const float* bl    = (const float*)d_in[13];
  const float* gam_l = (const float*)d_in[14];
  const float* bet_l = (const float*)d_in[15];
  const float* Wout  = (const float*)d_in[16];
  const float* bout  = (const float*)d_in[17];

  const int N   = in_sizes[2];           // 50000
  const int E   = in_sizes[1] / 2;       // 640000
  const int DIN = in_sizes[0] / N;       // 64
  const int GB  = in_sizes[5];           // 128
  const int GD  = in_sizes[7] / GB;      // 4
  const int LB  = in_sizes[11];          // 256
  const int LD  = in_sizes[13] / LB;     // 2
  const int G   = out_size / 2;          // 1024

  const int* src = eidx;
  const int* dst = eidx + E;

  // workspace carving (256B aligned)
  char* ws = (char*)d_ws;
  size_t off = 0;
  auto take = [&](size_t bytes) {
    char* p = ws + off;
    off = (off + bytes + 255) & ~(size_t)255;
    return p;
  };
  float*    dinv  = (float*)take((size_t)N * 4);
  float*    bufA  = (float*)take((size_t)N * GB * 4);
  float*    bufB  = (float*)take((size_t)N * GB * 4);
  float*    mA    = (float*)take((size_t)G * LB * 4);
  float*    mB    = (float*)take((size_t)G * LB * 4);
  float*    stats = (float*)take((size_t)2 * LB * 4);
  _Float16* hH    = (_Float16*)take((size_t)N * GB * 2);     // f16 GEMM input staging
  _Float16* w0T   = (_Float16*)take((size_t)DIN * GB * 2);   // transposed f16 weights
  _Float16* wgT   = (_Float16*)take((size_t)GD * GB * GB * 2);
  _Float16* wl0T  = (_Float16*)take((size_t)GB * LB * 2);
  _Float16* wlT   = (_Float16*)take((size_t)LD * LB * LB * 2);

  auto cdiv = [](size_t a, size_t b) { return (unsigned)((a + b - 1) / b); };
  dim3 gblk(32, 8);  // 8 wave32 waves -> 16 rows x 128 cols per block

  // ---- one-time (per launch) weight transpose + f16 convert ----
  k_cvt_wT<<<cdiv((size_t)DIN * GB, TPB), TPB, 0, stream>>>(W0, w0T, DIN, GB);
  for (int i = 0; i < GD; ++i)
    k_cvt_wT<<<cdiv((size_t)GB * GB, TPB), TPB, 0, stream>>>(
        Wg + (size_t)i * GB * GB, wgT + (size_t)i * GB * GB, GB, GB);
  k_cvt_wT<<<cdiv((size_t)GB * LB, TPB), TPB, 0, stream>>>(Wl0, wl0T, GB, LB);
  for (int i = 0; i < LD; ++i)
    k_cvt_wT<<<cdiv((size_t)LB * LB, TPB), TPB, 0, stream>>>(
        Wl + (size_t)i * LB * LB, wlT + (size_t)i * LB * LB, LB, LB);

  // ---- degree (with self loop) -> dinv = deg^{-1/2} ----
  k_fill<<<cdiv(N, TPB), TPB, 0, stream>>>(dinv, 1.0f, (size_t)N);
  k_deg_accum<<<cdiv(E, TPB), TPB, 0, stream>>>(dst, dinv, E);
  k_rsqrt_inplace<<<cdiv(N, TPB), TPB, 0, stream>>>(dinv, N);

  const size_t NG = (size_t)N * GB;

  // ---- GCN layer 0: relu(gcn(x, W0, b0)) ----
  k_cvt_f16<<<cdiv((size_t)N * DIN, TPB), TPB, 0, stream>>>(x, hH, (size_t)N * DIN);
  k_gemm_wmma<<<dim3(N / 16, GB / 128), gblk, 0, stream>>>(hH, w0T, nullptr, bufB, DIN, GB);
  k_fill<<<cdiv(NG, TPB), TPB, 0, stream>>>(bufA, 0.0f, NG);
  k_edge_scatter<<<cdiv((size_t)E * (GB / 4), TPB), TPB, 0, stream>>>(bufB, src, dst, dinv, bufA, E, GB);
  k_node_update<<<cdiv(NG, TPB), TPB, 0, stream>>>(bufA, bufB, dinv, b0, N, GB);
  k_relu<<<cdiv(NG, TPB), TPB, 0, stream>>>(bufA, NG);

  // ---- GCN layers 1..GD: relu(bn(gcn(h, Wg[i], bg[i]))) ----
  for (int i = 0; i < GD; ++i) {
    k_cvt_f16<<<cdiv(NG, TPB), TPB, 0, stream>>>(bufA, hH, NG);
    k_gemm_wmma<<<dim3(N / 16, GB / 128), gblk, 0, stream>>>(
        hH, wgT + (size_t)i * GB * GB, nullptr, bufB, GB, GB);
    k_fill<<<cdiv(NG, TPB), TPB, 0, stream>>>(bufA, 0.0f, NG);
    k_edge_scatter<<<cdiv((size_t)E * (GB / 4), TPB), TPB, 0, stream>>>(bufB, src, dst, dinv, bufA, E, GB);
    k_node_update<<<cdiv(NG, TPB), TPB, 0, stream>>>(bufA, bufB, dinv, bg + (size_t)i * GB, N, GB);
    k_fill<<<cdiv((size_t)2 * GB, TPB), TPB, 0, stream>>>(stats, 0.0f, (size_t)2 * GB);
    k_colstats<<<cdiv(N, 64), GB, 0, stream>>>(bufA, stats, N, GB);
    k_bn_relu<<<cdiv(NG, TPB), TPB, 0, stream>>>(bufA, stats, gam_g + (size_t)i * GB,
                                                 bet_g + (size_t)i * GB, N, GB);
  }

  // ---- global_add_pool ----
  k_fill<<<cdiv((size_t)G * GB, TPB), TPB, 0, stream>>>(mA, 0.0f, (size_t)G * GB);
  k_pool<<<cdiv(NG, TPB), TPB, 0, stream>>>(bufA, batch, mA, N, GB);

  // ---- MLP head ----
  const size_t GL = (size_t)G * LB;
  k_cvt_f16<<<cdiv((size_t)G * GB, TPB), TPB, 0, stream>>>(mA, hH, (size_t)G * GB);
  k_gemm_wmma<<<dim3(G / 16, LB / 128), gblk, 0, stream>>>(hH, wl0T, bl0, mB, GB, LB);
  k_relu<<<cdiv(GL, TPB), TPB, 0, stream>>>(mB, GL);

  float* cur = mB;
  float* oth = mA;
  for (int i = 0; i < LD; ++i) {
    k_cvt_f16<<<cdiv(GL, TPB), TPB, 0, stream>>>(cur, hH, GL);
    k_gemm_wmma<<<dim3(G / 16, LB / 128), gblk, 0, stream>>>(
        hH, wlT + (size_t)i * LB * LB, bl + (size_t)i * LB, oth, LB, LB);
    k_fill<<<cdiv((size_t)2 * LB, TPB), TPB, 0, stream>>>(stats, 0.0f, (size_t)2 * LB);
    k_colstats<<<cdiv(G, 64), LB, 0, stream>>>(oth, stats, G, LB);
    k_bn_relu<<<cdiv(GL, TPB), TPB, 0, stream>>>(oth, stats, gam_l + (size_t)i * LB,
                                                 bet_l + (size_t)i * LB, G, LB);
    float* t = cur; cur = oth; oth = t;
  }

  k_out<<<cdiv((size_t)G * 2, TPB), TPB, 0, stream>>>(cur, Wout, bout, (float*)d_out, G, LB);
}
